// CrossAttentionAudioFuser_89292370084405
// MI455X (gfx1250) — matrix operand order repaired and checked by reading, not compiled, not run on hardware
//
#include <hip/hip_runtime.h>
#include <hip/hip_fp16.h>

typedef __attribute__((ext_vector_type(16))) _Float16 v16h;
typedef __attribute__((ext_vector_type(4)))  _Float16 v4h;
typedef __attribute__((ext_vector_type(8)))  float    v8f;

#define B_SZ    16
#define C_IMG   512
#define HW      1024
#define C_AUD   512
#define K_LEN   256
#define N_HEADS 8
#define HEAD_D  64
#define SCALE   0.125f   // 64^-0.5 (folded into Wq/bq f16 conversion)
#define LN_EPS  1e-5f

// ---------------------------------------------------------------------------
// Fragment loaders matching CDNA5 wave32 WMMA VGPR layouts (ISA 7.12.2).
// A (16x32 f16): lane l: M = l%16 ; halves e=0..7 -> K = kBase + 8*(l/16) + e
//                              e=8..15 -> K = kBase + 16 + 8*(l/16) + (e-8)
// B (32x16 f16): lane l: N = nBase + l%16 ; K same as A (table order)
// C/D (16x16 f32): reg r: M = r + 8*(l/16) ; N = l%16
// ---------------------------------------------------------------------------
__device__ __forceinline__ v16h load_a16x32(const _Float16* __restrict__ rowBase,
                                            int ld, int kBase) {
    const int lane = threadIdx.x & 31;
    const int l16 = lane & 15, lg = lane >> 4;
    const _Float16* p = rowBase + (size_t)l16 * ld + kBase + 8 * lg;
    v16h a;
    *reinterpret_cast<int4*>(&a)       = *reinterpret_cast<const int4*>(p);
    *(reinterpret_cast<int4*>(&a) + 1) = *reinterpret_cast<const int4*>(p + 16);
    return a;
}

__device__ __forceinline__ v16h load_b32x16(const _Float16* __restrict__ src,
                                            int ld, int kBase, int nBase) {
    const int lane = threadIdx.x & 31;
    const int l16 = lane & 15, lg = lane >> 4;
    const _Float16* p = src + (size_t)(kBase + 8 * lg) * ld + nBase + l16;
    v16h b;
    #pragma unroll
    for (int e = 0; e < 8; ++e) {
        b[e] = p[(size_t)e * ld];
        b[8 + e] = p[(size_t)(16 + e) * ld];
    }
    return b;
}

__device__ __forceinline__ v8f wmma_f16(v16h a, v16h b, v8f c) {
    return __builtin_amdgcn_wmma_f32_16x16x32_f16(false, a, false, b,
                                                  (short)0, c, false, false);
}

// ---------------------------------------------------------------------------
// Cross-lane xor-butterfly within each 16-lane row, on the VALU
// (v_permlane16_b32 with constant nibble selectors) instead of ds_bpermute.
// ---------------------------------------------------------------------------
#if __has_builtin(__builtin_amdgcn_permlane16)
#define HAVE_PERMLANE16 1
__device__ __forceinline__ float permxor16(float v, unsigned s0, unsigned s1) {
    unsigned u = __builtin_bit_cast(unsigned, v);
    u = __builtin_amdgcn_permlane16(u, u, s0, s1, false, false);
    return __builtin_bit_cast(float, u);
}
#endif

__device__ __forceinline__ float redmax16(float v) {
#ifdef HAVE_PERMLANE16
    v = fmaxf(v, permxor16(v, 0x67452301u, 0xEFCDAB89u));  // xor 1
    v = fmaxf(v, permxor16(v, 0x54761032u, 0xDCFE98BAu));  // xor 2
    v = fmaxf(v, permxor16(v, 0x32107654u, 0xBA98FEDCu));  // xor 4
    v = fmaxf(v, permxor16(v, 0xFEDCBA98u, 0x76543210u));  // xor 8
#else
    #pragma unroll
    for (int m = 1; m < 16; m <<= 1) v = fmaxf(v, __shfl_xor(v, m, 32));
#endif
    return v;
}
__device__ __forceinline__ float redsum16(float v) {
#ifdef HAVE_PERMLANE16
    v += permxor16(v, 0x67452301u, 0xEFCDAB89u);
    v += permxor16(v, 0x54761032u, 0xDCFE98BAu);
    v += permxor16(v, 0x32107654u, 0xBA98FEDCu);
    v += permxor16(v, 0xFEDCBA98u, 0x76543210u);
#else
    #pragma unroll
    for (int m = 1; m < 16; m <<= 1) v += __shfl_xor(v, m, 32);
#endif
    return v;
}

// ---------------------------------------------------------------------------
// 1) vectorized f32 -> f16 (with optional scale folded in)
// ---------------------------------------------------------------------------
__global__ void __launch_bounds__(256)
cvt_f32_f16x4(const float* __restrict__ src,
              _Float16* __restrict__ dst, int n4, float scale) {
    int i = blockIdx.x * blockDim.x + threadIdx.x;
    if (i < n4) {
        float4 f = reinterpret_cast<const float4*>(src)[i];
        v4h h;
        h[0] = (_Float16)(f.x * scale);
        h[1] = (_Float16)(f.y * scale);
        h[2] = (_Float16)(f.z * scale);
        h[3] = (_Float16)(f.w * scale);
        reinterpret_cast<v4h*>(dst)[i] = h;
    }
}

// img_feat [B, C, HW] f32 -> img_seq [B, HW, C] f16
__global__ void __launch_bounds__(256)
img_to_seq(const float* __restrict__ img, _Float16* __restrict__ dst) {
    int i = blockIdx.x * blockDim.x + threadIdx.x;   // B*C*HW = 2^23
    int s = i & (HW - 1);
    int c = (i >> 10) & (C_IMG - 1);
    int b = i >> 19;
    dst[((size_t)b * HW + s) * C_IMG + c] = (_Float16)img[i];
}

// ---------------------------------------------------------------------------
// 2) Projection GEMM: out = A[M,512] @ W[512,512] + bias*bscale, f16 in/out.
// grid.x = M/16, grid.y = 4 (128-col bands), block = 256 (8 waves x 16 cols).
// MODE 0: row-major f16 out.  MODE 1: K pre-transposed per head kT[b,h,d,t].
// ---------------------------------------------------------------------------
template <int MODE>
__global__ void __launch_bounds__(256)
proj_gemm(const _Float16* __restrict__ A,
          const _Float16* __restrict__ W,
          const float* __restrict__ bias, float bscale,
          _Float16* __restrict__ out) {
    const int wave = threadIdx.x >> 5, lane = threadIdx.x & 31;
    const int l16 = lane & 15, lg = lane >> 4;
    const int row0 = blockIdx.x * 16;
    const int n0 = blockIdx.y * 128 + wave * 16;

    v8f acc;
    #pragma unroll
    for (int r = 0; r < 8; ++r) acc[r] = 0.0f;

    const _Float16* Arow = A + (size_t)row0 * 512;
    #pragma unroll 4
    for (int kc = 0; kc < 16; ++kc) {
        v16h a = load_a16x32(Arow, 512, 32 * kc);
        v16h b = load_b32x16(W, 512, 32 * kc, n0);
        acc = wmma_f16(a, b, acc);
    }
    const float bb = bias[n0 + l16] * bscale;
    #pragma unroll
    for (int r = 0; r < 8; ++r) {
        const int M = r + 8 * lg;
        const int row = row0 + M;
        const float val = acc[r] + bb;
        if (MODE == 0) {
            out[(size_t)row * 512 + n0 + l16] = (_Float16)val;
        } else {  // kT[b][h][d][t]
            const int ba = row >> 8, t = row & 255;
            const int col = n0 + l16, hh = col >> 6, d = col & 63;
            out[(((size_t)ba * N_HEADS + hh) * HEAD_D + d) * K_LEN + t] =
                (_Float16)val;
        }
    }
}

// ---------------------------------------------------------------------------
// 3) Attention: one wave = one (b, head, 16-row q-tile).
//    Full 16x256 score strip lives in 16 v8f accumulators; exact two-pass
//    softmax (one max-reduce + one sum-reduce per row), then P@V through a
//    1 KB per-wave LDS tile that re-layouts C/D -> A fragments.
//    __launch_bounds__(256, 1): min 1 wave/EU so the allocator may use the
//    full VGPR file (peak live set ~220 regs) without spilling to scratch.
// ---------------------------------------------------------------------------
__global__ void __launch_bounds__(256, 1)
attn_kernel(const _Float16* __restrict__ q,
            const _Float16* __restrict__ kT,
            const _Float16* __restrict__ v,
            _Float16* __restrict__ attn) {
    __shared__ __align__(16) _Float16 lds_p[8][16][32];   // 8 KB

    const int wave = threadIdx.x >> 5, lane = threadIdx.x & 31;
    const int l16 = lane & 15, lg = lane >> 4;
    const int task = blockIdx.x * 8 + wave;
    const int q0 = (task & 63) * 16;
    const int h = (task >> 6) & 7;
    const int b = task >> 9;

    const _Float16* qrow = q + ((size_t)b * HW + q0) * C_AUD + h * HEAD_D;
    const v16h aq0 = load_a16x32(qrow, C_AUD, 0);
    const v16h aq1 = load_a16x32(qrow, C_AUD, 32);
    const _Float16* kTh = kT + ((size_t)b * N_HEADS + h) * HEAD_D * K_LEN;
    const _Float16* vb = v + (size_t)b * K_LEN * C_AUD + h * HEAD_D;

    // --- scores: S = (q/8) @ K^T, 16 key tiles of 16 ---
    v8f s[16];
    #pragma unroll
    for (int t = 0; t < 16; ++t) {
        v8f z;
        #pragma unroll
        for (int r = 0; r < 8; ++r) z[r] = 0.0f;
        v16h bk0 = load_b32x16(kTh, K_LEN, 0, 16 * t);
        v16h bk1 = load_b32x16(kTh, K_LEN, 32, 16 * t);
        z = wmma_f16(aq0, bk0, z);
        s[t] = wmma_f16(aq1, bk1, z);
    }

    // --- exact softmax over 256 keys per row ---
    float inv[8];
    #pragma unroll
    for (int r = 0; r < 8; ++r) {
        float lm = s[0][r];
        #pragma unroll
        for (int t = 1; t < 16; ++t) lm = fmaxf(lm, s[t][r]);
        const float rm = redmax16(lm);
        float ls = 0.0f;
        #pragma unroll
        for (int t = 0; t < 16; ++t) {
            const float e = __expf(s[t][r] - rm);
            s[t][r] = e;
            ls += e;
        }
        inv[r] = 1.0f / redsum16(ls);
    }

    // --- O = P @ V, 32-key chunks via LDS re-layout ---
    v8f o[4];
    #pragma unroll
    for (int nt = 0; nt < 4; ++nt)
        #pragma unroll
        for (int r = 0; r < 8; ++r) o[nt][r] = 0.0f;

    #pragma unroll
    for (int kc = 0; kc < 8; ++kc) {
        #pragma unroll
        for (int r = 0; r < 8; ++r) {
            const int M = r + 8 * lg;
            lds_p[wave][M][l16]      = (_Float16)s[2 * kc][r];
            lds_p[wave][M][16 + l16] = (_Float16)s[2 * kc + 1][r];
        }
        // same-wave DS ops are in-order; re-read as A fragment
        const _Float16* pb = &lds_p[wave][0][0];
        v16h ap;
        *reinterpret_cast<int4*>(&ap) =
            *reinterpret_cast<const int4*>(pb + l16 * 32 + 8 * lg);
        *(reinterpret_cast<int4*>(&ap) + 1) =
            *reinterpret_cast<const int4*>(pb + l16 * 32 + 16 + 8 * lg);

        #pragma unroll
        for (int nt = 0; nt < 4; ++nt) {
            v16h bv = load_b32x16(vb, C_AUD, 32 * kc, 16 * nt);
            o[nt] = wmma_f16(ap, bv, o[nt]);
        }
    }

    #pragma unroll
    for (int r = 0; r < 8; ++r) {
        const int M = r + 8 * lg;
        _Float16* orow = attn + ((size_t)b * HW + q0 + M) * C_AUD + h * HEAD_D;
        #pragma unroll
        for (int nt = 0; nt < 4; ++nt)
            orow[16 * nt + l16] = (_Float16)(o[nt][r] * inv[r]);
    }
}

// ---------------------------------------------------------------------------
// 4) O-projection + bias + residual + LayerNorm, output NCHW f32.
//    Block = 16 rows; 8 waves x 64 cols cover C=512.
// ---------------------------------------------------------------------------
__global__ void __launch_bounds__(256, 1)
oproj_ln(const _Float16* __restrict__ attn,
         const _Float16* __restrict__ Wo,
         const float* __restrict__ bo,
         const float* __restrict__ img,
         const float* __restrict__ gamma,
         const float* __restrict__ beta,
         float* __restrict__ out) {
    __shared__ float rsum[16], rsq[16];
    const int wave = threadIdx.x >> 5, lane = threadIdx.x & 31;
    const int l16 = lane & 15, lg = lane >> 4;
    const int row0 = blockIdx.x * 16;

    if (threadIdx.x < 16) { rsum[threadIdx.x] = 0.0f; rsq[threadIdx.x] = 0.0f; }
    __syncthreads();

    v8f acc[4];
    #pragma unroll
    for (int nt = 0; nt < 4; ++nt)
        #pragma unroll
        for (int r = 0; r < 8; ++r) acc[nt][r] = 0.0f;

    const _Float16* Arow = attn + (size_t)row0 * 512;
    #pragma unroll 2
    for (int kc = 0; kc < 16; ++kc) {
        v16h a = load_a16x32(Arow, 512, 32 * kc);
        #pragma unroll
        for (int nt = 0; nt < 4; ++nt) {
            v16h bw = load_b32x16(Wo, 512, 32 * kc, wave * 64 + 16 * nt);
            acc[nt] = wmma_f16(a, bw, acc[nt]);
        }
    }

    float y[4][8], ps[8], ps2[8];
    #pragma unroll
    for (int r = 0; r < 8; ++r) { ps[r] = 0.0f; ps2[r] = 0.0f; }

    #pragma unroll
    for (int nt = 0; nt < 4; ++nt) {
        const int c = wave * 64 + 16 * nt + l16;
        const float bb = bo[c];
        #pragma unroll
        for (int r = 0; r < 8; ++r) {
            const int M = r + 8 * lg;
            const int grow = row0 + M;
            const int b = grow >> 10, s = grow & 1023;
            const float val =
                acc[nt][r] + bb + img[((size_t)b * C_IMG + c) * HW + s];
            y[nt][r] = val;
            ps[r] += val;
            ps2[r] += val * val;
        }
    }
    #pragma unroll
    for (int r = 0; r < 8; ++r) {
        const float a1 = redsum16(ps[r]);
        const float a2 = redsum16(ps2[r]);
        if (l16 == 0) {
            const int M = r + 8 * lg;
            atomicAdd(&rsum[M], a1);
            atomicAdd(&rsq[M], a2);
        }
    }
    __syncthreads();

    #pragma unroll
    for (int nt = 0; nt < 4; ++nt) {
        const int c = wave * 64 + 16 * nt + l16;
        const float g = gamma[c], be = beta[c];
        #pragma unroll
        for (int r = 0; r < 8; ++r) {
            const int M = r + 8 * lg;
            const float mu = rsum[M] * (1.0f / 512.0f);
            const float var = rsq[M] * (1.0f / 512.0f) - mu * mu;
            const float rsig = rsqrtf(var + LN_EPS);
            const int grow = row0 + M;
            const int b = grow >> 10, s = grow & 1023;
            out[((size_t)b * C_IMG + c) * HW + s] =
                (y[nt][r] - mu) * rsig * g + be;
        }
    }
}

// ---------------------------------------------------------------------------
// Host launcher
// ---------------------------------------------------------------------------
extern "C" void kernel_launch(void* const* d_in, const int* in_sizes, int n_in,
                              void* d_out, int out_size, void* d_ws,
                              size_t ws_size, hipStream_t stream) {
    (void)in_sizes; (void)n_in; (void)out_size; (void)ws_size;
    const float* img   = (const float*)d_in[0];
    const float* aud   = (const float*)d_in[1];
    const float* Wq    = (const float*)d_in[2];
    const float* bq    = (const float*)d_in[3];
    const float* Wk    = (const float*)d_in[4];
    const float* bk    = (const float*)d_in[5];
    const float* Wv    = (const float*)d_in[6];
    const float* bv    = (const float*)d_in[7];
    const float* Wo    = (const float*)d_in[8];
    const float* bo    = (const float*)d_in[9];
    const float* gamma = (const float*)d_in[10];
    const float* beta  = (const float*)d_in[11];
    float* out = (float*)d_out;

    char* ws = (char*)d_ws;
    const size_t W_ELEMS = 512 * 512;                 // 262144
    _Float16* wq_h  = (_Float16*)(ws);                          // 0.5 MB each
    _Float16* wk_h  = (_Float16*)(ws + (1 * W_ELEMS) * 2);
    _Float16* wv_h  = (_Float16*)(ws + (2 * W_ELEMS) * 2);
    _Float16* wo_h  = (_Float16*)(ws + (3 * W_ELEMS) * 2);
    size_t off = 4 * W_ELEMS * 2;                     // 2 MB
    _Float16* aud_h = (_Float16*)(ws + off); off += (size_t)B_SZ * K_LEN * C_AUD * 2; // +4 MB
    _Float16* img_h = (_Float16*)(ws + off); off += (size_t)B_SZ * HW * C_IMG * 2;    // +16 MB
    _Float16* q_h   = (_Float16*)(ws + off); off += (size_t)B_SZ * HW * C_AUD * 2;    // +16 MB
    _Float16* kT_h  = (_Float16*)(ws + off); off += (size_t)B_SZ * N_HEADS * HEAD_D * K_LEN * 2; // +4 MB
    _Float16* v_h   = (_Float16*)(ws + off); off += (size_t)B_SZ * K_LEN * C_AUD * 2; // +4 MB
    _Float16* at_h  = (_Float16*)(ws + off);                                          // +16 MB

    const int T = 256;
    // weights + audio -> f16 (Wq pre-scaled by 1/sqrt(d))
    cvt_f32_f16x4<<<(int)(W_ELEMS / 4 / T), T, 0, stream>>>(Wq, wq_h, (int)(W_ELEMS / 4), SCALE);
    cvt_f32_f16x4<<<(int)(W_ELEMS / 4 / T), T, 0, stream>>>(Wk, wk_h, (int)(W_ELEMS / 4), 1.0f);
    cvt_f32_f16x4<<<(int)(W_ELEMS / 4 / T), T, 0, stream>>>(Wv, wv_h, (int)(W_ELEMS / 4), 1.0f);
    cvt_f32_f16x4<<<(int)(W_ELEMS / 4 / T), T, 0, stream>>>(Wo, wo_h, (int)(W_ELEMS / 4), 1.0f);
    const int AUD_N = B_SZ * K_LEN * C_AUD;
    cvt_f32_f16x4<<<AUD_N / 4 / T, T, 0, stream>>>(aud, aud_h, AUD_N / 4, 1.0f);
    // img NCHW -> [B, HW, C] f16
    img_to_seq<<<(B_SZ * C_IMG * HW) / T, T, 0, stream>>>(img, img_h);

    // projections (q gets bias scaled by 1/8 to match pre-scaled Wq)
    proj_gemm<0><<<dim3(B_SZ * HW / 16, 4), T, 0, stream>>>(img_h, wq_h, bq, SCALE, q_h);
    proj_gemm<1><<<dim3(B_SZ * K_LEN / 16, 4), T, 0, stream>>>(aud_h, wk_h, bk, 1.0f, kT_h);
    proj_gemm<0><<<dim3(B_SZ * K_LEN / 16, 4), T, 0, stream>>>(aud_h, wv_h, bv, 1.0f, v_h);

    // attention: B*heads*(HW/16) wave-tasks, 8 per block
    attn_kernel<<<(B_SZ * N_HEADS * (HW / 16)) / 8, T, 0, stream>>>(q_h, kT_h, v_h, at_h);

    // output projection + residual + layernorm
    oproj_ln<<<B_SZ * HW / 16, T, 0, stream>>>(at_h, wo_h, bo, img, gamma, beta, out);
}
